// MultiHeadAttention_82523501626048
// MI455X (gfx1250) — matrix operand order repaired and checked by reading, not compiled, not run on hardware
//
#include <hip/hip_runtime.h>

// ---------------------------------------------------------------------------
// MultiHeadAttention forward for MI455X (gfx1250, wave32, WMMA)
// B=2, S=2048, E=1024, H=16, DK=DV=64
//
// Stage 0: convert activations f32->bf16 (once per element)
//          transpose+convert weights f32 [K][N] -> bf16 [N][K]
// Stage 1: 3x projection GEMM, all-bf16 operands, async-LDS staged
//          (GLOBAL_LOAD_ASYNC_TO_LDS_B128), double-buffered, K-slab 64,
//          v_wmma_f32_16x16x32_bf16
// Stage 2: flash attention (bf16 WMMA, f32 online softmax), ctx -> bf16
// Stage 3: output GEMM -> f32 d_out
// ---------------------------------------------------------------------------

typedef __bf16 bf16_t;
typedef __attribute__((ext_vector_type(16))) __bf16 v16bf;
typedef __attribute__((ext_vector_type(8)))  __bf16 v8bf;
typedef __attribute__((ext_vector_type(4)))  __bf16 v4bf;
typedef __attribute__((ext_vector_type(2)))  __bf16 v2bf;
typedef __attribute__((ext_vector_type(8)))  float  v8f;

#define BB 2
#define SS 2048
#define EE 1024
#define HH 16
#define DD 64
#define KT 64                 // K-slab per staged LDS tile
#define KITER (EE / KT)       // 16
#define NEGBIG (-1e9f)

static __device__ __forceinline__ bf16_t f2bf(float f) {
    // round-to-nearest-even f32 -> bf16
    unsigned u = __builtin_bit_cast(unsigned, f);
    u += 0x7fffu + ((u >> 16) & 1u);
    unsigned short s = (unsigned short)(u >> 16);
    return __builtin_bit_cast(bf16_t, s);
}

static __device__ __forceinline__ v2bf cvt2(float a, float b) {
    v2bf r; r[0] = f2bf(a); r[1] = f2bf(b); return r;
}

static __device__ __forceinline__ v8bf cvt8(float4 a, float4 b) {
    v2bf p0 = cvt2(a.x, a.y), p1 = cvt2(a.z, a.w);
    v2bf p2 = cvt2(b.x, b.y), p3 = cvt2(b.z, b.w);
    v4bf q0 = __builtin_shufflevector(p0, p1, 0, 1, 2, 3);
    v4bf q1 = __builtin_shufflevector(p2, p3, 0, 1, 2, 3);
    return __builtin_shufflevector(q0, q1, 0, 1, 2, 3, 4, 5, 6, 7);
}

static __device__ __forceinline__ v16bf cat8(v8bf lo, v8bf hi) {
    return __builtin_shufflevector(lo, hi, 0,1,2,3,4,5,6,7,8,9,10,11,12,13,14,15);
}

static __device__ __forceinline__ v8f wmma_bf16(v16bf a, v16bf b, v8f c) {
    // D(16x16 f32) = A(16x32 bf16) x B(32x16 bf16) + C
    return __builtin_amdgcn_wmma_f32_16x16x32_bf16(false, a, false, b,
                                                   (short)0, c, false, false);
}

// Async copy of one 16-byte chunk global -> LDS (CDNA5 GLOBAL_LOAD_ASYNC_TO_LDS,
// ISA 15.18.3 op 98). LDS byte offset = low 32 bits of the generic pointer
// (ISA 10.2). Tracked by ASYNCcnt.
static __device__ __forceinline__ void async_cp16(const bf16_t* g, bf16_t* l) {
    unsigned lds = (unsigned)(unsigned long long)l;
    unsigned long long ga = (unsigned long long)g;
    asm volatile("global_load_async_to_lds_b128 %0, %1, off"
                 :: "v"(lds), "v"(ga) : "memory");
}

static __device__ __forceinline__ void wait_async0() {
    asm volatile("s_wait_asynccnt 0x0" ::: "memory");
}

// ---------------------------------------------------------------------------
// Elementwise f32 -> bf16 (each element converted exactly once)
// ---------------------------------------------------------------------------
__global__ __launch_bounds__(256)
void cvt_f32_bf16_kernel(const float* __restrict__ in, bf16_t* __restrict__ out)
{
    const size_t i = ((size_t)blockIdx.x * 256 + threadIdx.x) * 8;
    const float4* p = (const float4*)(in + i);
    float4 a = p[0], b = p[1];
    *(v8bf*)(out + i) = cvt8(a, b);
}

// ---------------------------------------------------------------------------
// Transpose + convert: W f32 [1024][1024] (k-major) -> WT bf16 [n][k]
// 32x32 tiles through LDS (padded), pair-packed bf16 stores.
// ---------------------------------------------------------------------------
__global__ __launch_bounds__(256)
void transpose_cvt_kernel(const float* __restrict__ W, bf16_t* __restrict__ WT)
{
    __shared__ float tile[32][33];
    const int tid = threadIdx.x;
    const int k0 = blockIdx.y * 32, n0 = blockIdx.x * 32;

    const int tx = tid & 31, ty = tid >> 5;          // 32 x 8
#pragma unroll
    for (int r = 0; r < 4; ++r)
        tile[ty + r * 8][tx] = W[(size_t)(k0 + ty + r * 8) * EE + n0 + tx];
    __syncthreads();

    const int c = (tid & 15) * 2, nr = tid >> 4;     // 16 x 16
#pragma unroll
    for (int r = 0; r < 2; ++r) {
        const int n = nr + r * 16;
        v2bf p = cvt2(tile[c][n], tile[c + 1][n]);
        *(v2bf*)(&WT[(size_t)(n0 + n) * EE + k0 + c]) = p;
    }
}

// ---------------------------------------------------------------------------
// All-bf16 GEMM: C[M=4096, N=1024] = A[M,1024] @ WT[N,1024]^T + bias
// 256 threads = 8 waves in a 4(M) x 2(N) grid; WG tile 128x128; K-slab 64.
// Both LDS tiles staged by async global->LDS copies, double-buffered.
// One s_wait_asynccnt + one barrier per 16 WMMAs. #pragma unroll 1 keeps
// accumulators in a single stable allocation (no unroll register rotation).
// mode 0: outF[row*1024+col] = val              (f32, final projection)
// mode 1: outB[b][h][s][d]  = bf16(val)         (Q / K layout)
// mode 2: outB[b][h][d][s]  = bf16(val)         (V transposed)
// ---------------------------------------------------------------------------
__global__ __launch_bounds__(256)
void gemm_bf16_kernel(const bf16_t* __restrict__ A, const bf16_t* __restrict__ WT,
                      const float* __restrict__ bias,
                      float* __restrict__ outF, bf16_t* __restrict__ outB,
                      int mode)
{
    __shared__ __align__(64) bf16_t As[2][128][KT];   // [buf][m][k]  32 KB
    __shared__ __align__(64) bf16_t Bs[2][128][KT];   // [buf][n][k]  32 KB

    const int tid  = threadIdx.x;
    const int wave = tid >> 5;
    const int lane = tid & 31;
    const int half = lane >> 4;          // 0: lanes 0-15, 1: lanes 16-31
    const int ml   = lane & 15;
    const int wm   = wave >> 1;          // 0..3 : 32-row strip
    const int wn   = wave & 1;           // 0..1 : 64-col strip
    const int m0   = blockIdx.y * 128;
    const int n0   = blockIdx.x * 128;

    const v8f vzero = {0.f,0.f,0.f,0.f,0.f,0.f,0.f,0.f};
    v8f acc[2][4];
#pragma unroll
    for (int mt = 0; mt < 2; ++mt)
#pragma unroll
        for (int nt = 0; nt < 4; ++nt) acc[mt][nt] = vzero;

    // staging: 128 rows x 128B per tile; thread t copies a contiguous 64B run
    const int srow = tid >> 1;            // 0..127
    const int scol = (tid & 1) * 32;      // bf16 element offset 0 or 32

    const bf16_t* aP = A  + (size_t)(m0 + srow) * EE + scol;
    const bf16_t* wP = WT + (size_t)(n0 + srow) * EE + scol;

    auto stage = [&](int nb, int kElem) {
        const bf16_t* a = aP + kElem;
        const bf16_t* w = wP + kElem;
        bf16_t* la = &As[nb][srow][scol];
        bf16_t* lb = &Bs[nb][srow][scol];
        async_cp16(a,      la);
        async_cp16(a + 8,  la + 8);
        async_cp16(a + 16, la + 16);
        async_cp16(a + 24, la + 24);
        async_cp16(w,      lb);
        async_cp16(w + 8,  lb + 8);
        async_cp16(w + 16, lb + 16);
        async_cp16(w + 24, lb + 24);
    };

    // ---- prologue: async-stage k-slab 0 into buffer 0 ----
    stage(0, 0);

#pragma unroll 1
    for (int it = 0; it < KITER; ++it) {
        const int buf = it & 1;
        wait_async0();      // this wave's copies into `buf` have landed
        __syncthreads();    // all waves' copies have landed

        // ---- issue async staging of the next k-slab into buf^1 ----
        if (it + 1 < KITER) stage(buf ^ 1, (it + 1) * KT);

        // ---- fragments from `buf`; 16 WMMAs (2 kk x 2 A-frags x 4 B-frags) --
        v16bf af[2][2];
#pragma unroll
        for (int kk = 0; kk < 2; ++kk)
#pragma unroll
            for (int mt = 0; mt < 2; ++mt) {
                const bf16_t* r = &As[buf][wm * 32 + mt * 16 + ml][kk * 32];
                af[kk][mt] = *(const v16bf*)(r + 16 * half);
            }
#pragma unroll
        for (int kk = 0; kk < 2; ++kk)
#pragma unroll
            for (int nt = 0; nt < 4; ++nt) {
                v16bf bfrag = *(const v16bf*)
                    (&Bs[buf][wn * 64 + nt * 16 + ml][kk * 32 + 16 * half]);
                acc[0][nt] = wmma_bf16(af[kk][0], bfrag, acc[0][nt]);
                acc[1][nt] = wmma_bf16(af[kk][1], bfrag, acc[1][nt]);
            }
    }

    // ---- epilogue: C/D layout row = v + 8*half, col = nt*16 + ml ----
#pragma unroll
    for (int mt = 0; mt < 2; ++mt)
#pragma unroll
        for (int nt = 0; nt < 4; ++nt) {
            const int col = n0 + wn * 64 + nt * 16 + ml;
            const float bb = bias[col];
#pragma unroll
            for (int v = 0; v < 8; ++v) {
                const int row = m0 + wm * 32 + mt * 16 + v + 8 * half;
                const float val = acc[mt][nt][v] + bb;
                if (mode == 0) {
                    outF[(size_t)row * EE + col] = val;
                } else {
                    const int bi = row >> 11, s = row & (SS - 1);
                    const int hh = col >> 6,  d = col & (DD - 1);
                    if (mode == 1)
                        outB[(((size_t)bi * HH + hh) * SS + s) * DD + d] =
                            f2bf(val);
                    else
                        outB[(((size_t)bi * HH + hh) * DD + d) * SS + s] =
                            f2bf(val);
                }
            }
        }
}

// ---------------------------------------------------------------------------
// Flash attention: grid (S/128, H, B), 256 threads = 8 waves.
// Each wave owns 16 query rows; online softmax over kv blocks of 32.
// Q/K in [b,h,s,d] bf16; V transposed [b,h,d,s] bf16 so all K/V fragments
// are contiguous 32B v16bf global loads. P tile converted accum-layout ->
// A-fragment layout through wave-private LDS (ordered by s_wait_dscnt).
// Output ctx written as bf16 [b, s, h*64+d] (A-input of the final GEMM).
// ---------------------------------------------------------------------------
__global__ __launch_bounds__(256)
void flash_attn_kernel(const bf16_t* __restrict__ Q, const bf16_t* __restrict__ K,
                       const bf16_t* __restrict__ VT, const int* __restrict__ mask,
                       bf16_t* __restrict__ ctxB)
{
    __shared__ __align__(64) bf16_t pbuf[8][16][32];   // 8 KB, wave-private rows

    const int tid  = threadIdx.x;
    const int wave = tid >> 5;
    const int lane = tid & 31;
    const int half = lane >> 4;
    const int ml   = lane & 15;
    const int b    = blockIdx.z;
    const int hh   = blockIdx.y;
    const int qbase = blockIdx.x * 128 + wave * 16;

    const bf16_t* qp = Q  + ((size_t)b * HH + hh) * SS * DD;
    const bf16_t* kp = K  + ((size_t)b * HH + hh) * SS * DD;
    const bf16_t* vp = VT + ((size_t)b * HH + hh) * DD * SS;

    // Q A-fragments for K-steps d0=0,32 (loaded once; rows fixed per wave)
    v16bf qa[2];
#pragma unroll
    for (int kk = 0; kk < 2; ++kk) {
        const bf16_t* r = qp + (size_t)(qbase + ml) * DD + kk * 32;
        qa[kk] = *(const v16bf*)(r + 16 * half);
    }

    float Mx[8], Ls[8];
#pragma unroll
    for (int v = 0; v < 8; ++v) { Mx[v] = -1e30f; Ls[v] = 0.f; }
    const v8f vzero = {0.f,0.f,0.f,0.f,0.f,0.f,0.f,0.f};
    v8f cacc[4];
#pragma unroll
    for (int dt = 0; dt < 4; ++dt) cacc[dt] = vzero;

    for (int kv0 = 0; kv0 < SS; kv0 += 32) {
        if (kv0 + 32 < SS) {   // prefetch next kv block (K rows + mask row)
            __builtin_prefetch(kp + (size_t)(kv0 + 32 + ml) * DD, 0, 1);
            __builtin_prefetch(mask + ((size_t)b * SS + qbase + ml) * SS
                               + kv0 + 32, 0, 1);
        }

        // ---- scores: 16 x 32 tile = two 16x16 WMMAs, contract over d=64 ----
        v8f s0 = vzero, s1 = vzero;
#pragma unroll
        for (int kk = 0; kk < 2; ++kk) {
            v16bf kb0 = *(const v16bf*)(kp + (size_t)(kv0 + ml) * DD
                                        + kk * 32 + 16 * half);
            v16bf kb1 = *(const v16bf*)(kp + (size_t)(kv0 + 16 + ml) * DD
                                        + kk * 32 + 16 * half);
            s0 = wmma_bf16(qa[kk], kb0, s0);
            s1 = wmma_bf16(qa[kk], kb1, s1);
        }

        // ---- mask + scale (1/sqrt(64)=0.125); rowwise max ----
        float rm[8];
#pragma unroll
        for (int v = 0; v < 8; ++v) {
            const int row = qbase + v + 8 * half;
            const int* mrow = mask + ((size_t)b * SS + row) * SS + kv0 + ml;
            float x0 = s0[v] * 0.125f; if (mrow[0]  == 0) x0 = NEGBIG;
            float x1 = s1[v] * 0.125f; if (mrow[16] == 0) x1 = NEGBIG;
            s0[v] = x0; s1[v] = x1;
            rm[v] = fmaxf(x0, x1);
        }
#pragma unroll
        for (int v = 0; v < 8; ++v) {   // reduce across the 16-lane half
            rm[v] = fmaxf(rm[v], __shfl_xor(rm[v], 1, 32));
            rm[v] = fmaxf(rm[v], __shfl_xor(rm[v], 2, 32));
            rm[v] = fmaxf(rm[v], __shfl_xor(rm[v], 4, 32));
            rm[v] = fmaxf(rm[v], __shfl_xor(rm[v], 8, 32));
        }

        // ---- online softmax update; write P (bf16) to wave-private LDS ----
        float al[8], p0v[8], p1v[8];
#pragma unroll
        for (int v = 0; v < 8; ++v) {
            const float nm = fmaxf(Mx[v], rm[v]);
            al[v] = __expf(Mx[v] - nm);
            Mx[v] = nm;
            const float p0 = __expf(s0[v] - nm);
            const float p1 = __expf(s1[v] - nm);
            p0v[v] = p0; p1v[v] = p1;
            v2bf pp = cvt2(p0, p1);
            pbuf[wave][v + 8 * half][ml]      = pp[0];
            pbuf[wave][v + 8 * half][16 + ml] = pp[1];
        }
#pragma unroll
        for (int v = 0; v < 8; ++v) {
            float rs = p0v[v] + p1v[v];
            rs += __shfl_xor(rs, 1, 32);
            rs += __shfl_xor(rs, 2, 32);
            rs += __shfl_xor(rs, 4, 32);
            rs += __shfl_xor(rs, 8, 32);
            Ls[v] = Ls[v] * al[v] + rs;
        }
#pragma unroll
        for (int dt = 0; dt < 4; ++dt)
#pragma unroll
            for (int v = 0; v < 8; ++v) cacc[dt][v] *= al[v];

        // wave-internal LDS ordering (cross-lane visibility within the wave)
        asm volatile("s_wait_dscnt 0x0" ::: "memory");

        // ---- P A-fragment; ctx += P(16x32) @ V(32x64) : 4 WMMAs ----
        v16bf pa = *(const v16bf*)(&pbuf[wave][ml][16 * half]);
#pragma unroll
        for (int dt = 0; dt < 4; ++dt) {
            v16bf vb = *(const v16bf*)(vp + (size_t)(dt * 16 + ml) * SS
                                       + kv0 + 16 * half);
            cacc[dt] = wmma_bf16(pa, vb, cacc[dt]);
        }
    }

    // ---- normalize, store ctx bf16 in [b, s, h*64+d] for the final GEMM ----
#pragma unroll
    for (int dt = 0; dt < 4; ++dt)
#pragma unroll
        for (int v = 0; v < 8; ++v) {
            const int row = qbase + v + 8 * half;
            const float inv = 1.0f / Ls[v];
            ctxB[((size_t)b * SS + row) * (HH * DD) + hh * DD + dt * 16 + ml] =
                f2bf(cacc[dt][v] * inv);
        }
}

// ---------------------------------------------------------------------------
extern "C" void kernel_launch(void* const* d_in, const int* in_sizes, int n_in,
                              void* d_out, int out_size, void* d_ws, size_t ws_size,
                              hipStream_t stream)
{
    const float* query = (const float*)d_in[0];
    const float* key   = (const float*)d_in[1];
    const float* value = (const float*)d_in[2];
    const int*   mask  = (const int*)  d_in[3];
    const float* Wq    = (const float*)d_in[4];
    const float* bq    = (const float*)d_in[5];
    const float* Wk    = (const float*)d_in[6];
    const float* bk    = (const float*)d_in[7];
    const float* Wv    = (const float*)d_in[8];
    const float* bv    = (const float*)d_in[9];
    const float* Wo    = (const float*)d_in[10];
    const float* bo    = (const float*)d_in[11];
    float* out = (float*)d_out;

    char* ws = (char*)d_ws;
    const size_t nAct = (size_t)BB * SS * EE;            // 4M elements
    const size_t nW   = (size_t)EE * EE;                 // 1M elements
    size_t off = 0;
    bf16_t* xq  = (bf16_t*)(ws + off); off += nAct * 2;  //  8 MB
    bf16_t* xk  = (bf16_t*)(ws + off); off += nAct * 2;  //  8 MB
    bf16_t* xv  = (bf16_t*)(ws + off); off += nAct * 2;  //  8 MB
    bf16_t* wtq = (bf16_t*)(ws + off); off += nW * 2;    //  2 MB
    bf16_t* wtk = (bf16_t*)(ws + off); off += nW * 2;    //  2 MB
    bf16_t* wtv = (bf16_t*)(ws + off); off += nW * 2;    //  2 MB
    bf16_t* wto = (bf16_t*)(ws + off); off += nW * 2;    //  2 MB
    bf16_t* qws = (bf16_t*)(ws + off); off += nAct * 2;  //  8 MB [b,h,s,d]
    bf16_t* kws = (bf16_t*)(ws + off); off += nAct * 2;  //  8 MB [b,h,s,d]
    bf16_t* vws = (bf16_t*)(ws + off); off += nAct * 2;  //  8 MB [b,h,d,s]
    bf16_t* ctx = (bf16_t*)(ws + off); off += nAct * 2;  //  8 MB [b,s,h*d]

    // stage 0: one-time conversions
    const int cvtBlocks = (int)(nAct / (256 * 8));       // 2048
    cvt_f32_bf16_kernel<<<cvtBlocks, 256, 0, stream>>>(query, xq);
    cvt_f32_bf16_kernel<<<cvtBlocks, 256, 0, stream>>>(key,   xk);
    cvt_f32_bf16_kernel<<<cvtBlocks, 256, 0, stream>>>(value, xv);
    const dim3 tgrid(EE / 32, EE / 32);                  // (32, 32)
    transpose_cvt_kernel<<<tgrid, 256, 0, stream>>>(Wq, wtq);
    transpose_cvt_kernel<<<tgrid, 256, 0, stream>>>(Wk, wtk);
    transpose_cvt_kernel<<<tgrid, 256, 0, stream>>>(Wv, wtv);
    transpose_cvt_kernel<<<tgrid, 256, 0, stream>>>(Wo, wto);

    // stage 1: projections
    const dim3 gblk(EE / 128, (BB * SS) / 128, 1);       // (8, 32)
    gemm_bf16_kernel<<<gblk, 256, 0, stream>>>(xq, wtq, bq, nullptr, qws, 1);
    gemm_bf16_kernel<<<gblk, 256, 0, stream>>>(xk, wtk, bk, nullptr, kws, 1);
    gemm_bf16_kernel<<<gblk, 256, 0, stream>>>(xv, wtv, bv, nullptr, vws, 2);

    // stage 2: attention
    flash_attn_kernel<<<dim3(SS / 128, HH, BB), 256, 0, stream>>>(
        qws, kws, vws, mask, ctx);

    // stage 3: output projection -> f32
    gemm_bf16_kernel<<<gblk, 256, 0, stream>>>(ctx, wto, bo, out, nullptr, 0);
}
